// LightSpatialTransformerLayer_74071005986904
// MI455X (gfx1250) — matrix operand order repaired and checked by reading, not compiled, not run on hardware
//
#include <hip/hip_runtime.h>
#include <hip/hip_bf16.h>
#include <math.h>

#define BQ    4
#define NPT   4096
#define DCH   256
#define NSAMP 16

typedef __attribute__((ext_vector_type(16))) _Float16 v16h;
typedef __attribute__((ext_vector_type(8)))  float    v8f;
typedef _Float16 h16;

__device__ __forceinline__ float gelu_exact(float x) {
    return 0.5f * x * (1.0f + erff(x * 0.70710678118654752440f));
}

__device__ __forceinline__ v8f zero8() {
    v8f z;
#pragma unroll
    for (int i = 0; i < 8; ++i) z[i] = 0.0f;
    return z;
}

__device__ __forceinline__ v8f wmma_f16(v16h a, v16h b, v8f c) {
    // D = A(16x32 f16) * B(32x16 f16) + C(16x16 f32)
    return __builtin_amdgcn_wmma_f32_16x16x32_f16(false, a, false, b, (short)0, c,
                                                  false, false);
}

// A-fragment (16x32 f16) from f16 LDS, row-major [16][DCH], columns kBase..kBase+31.
// Lane l: row = l&15; lanes 0-15 hold K {0..7,16..23}, lanes 16-31 {8..15,24..31}
// (pairs of consecutive K packed per VGPR).
__device__ __forceinline__ v16h a_frag_f16(const h16* base, int kBase, int lane) {
    const int row = lane & 15, hi = lane >> 4;
    const h16* r = base + row * DCH + kBase + hi * 8;
    v16h a;
#pragma unroll
    for (int v = 0; v < 8; ++v) {
        const int K = ((v >> 2) << 4) + ((v & 3) << 1);
        a[2 * v]     = r[K];
        a[2 * v + 1] = r[K + 1];
    }
    return a;
}

// Same fragment, sourced from f32 LDS with on-the-fly convert.
__device__ __forceinline__ v16h a_frag_f32(const float* base, int kBase, int lane) {
    const int row = lane & 15, hi = lane >> 4;
    const float* r = base + row * DCH + kBase + hi * 8;
    v16h a;
#pragma unroll
    for (int v = 0; v < 8; ++v) {
        const int K = ((v >> 2) << 4) + ((v & 3) << 1);
        a[2 * v]     = (h16)r[K];
        a[2 * v + 1] = (h16)r[K + 1];
    }
    return a;
}

// ---------------------------------------------------------------------------
// Pack a row-major f32 weight [K x Ncol] into f16 WMMA B-fragment tiles:
// tile (kt,nt) holds B[kt*32 .. +31][nt*16 .. +15]; within a tile, lane l owns
// 16 consecutive halfs: col = l&15, K = (l>>4)*8 + (e&7) + (e>>3)*16. One 32B load per lane.
// ---------------------------------------------------------------------------
__global__ __launch_bounds__(256) void pack_weight(const float* __restrict__ W,
                                                   h16* __restrict__ out,
                                                   int K, int Ncol) {
    const int t = blockIdx.x * blockDim.x + threadIdx.x;
    if (t >= K * Ncol) return;
    const int tile = t >> 9;
    const int r    = t & 511;
    const int lane = r >> 4;
    const int e    = r & 15;
    const int ntiles = Ncol >> 4;
    const int kt = tile / ntiles, nt = tile - kt * ntiles;
    const int k = kt * 32 + (lane >> 4) * 8 + (e & 7) + (e >> 3) * 16;
    const int n = nt * 16 + (lane & 15);
    out[t] = (h16)W[k * Ncol + n];
}

// ---------------------------------------------------------------------------
// Kernel 1: per query point, kth = 8th-smallest distance, then the reference's
// stable-argsort selection: first 16 indices j with (j==0 || !in_box),
// falling back to ascending in-box indices. One wave per query point.
// ---------------------------------------------------------------------------
__global__ __launch_bounds__(256) void knn_box_kernel(const float* __restrict__ xyz,
                                                      int* __restrict__ idxout) {
    __shared__ float sx[NPT], sy[NPT], sz[NPT];
    const int b     = blockIdx.x >> 6;   // 64 chunks per batch
    const int chunk = blockIdx.x & 63;
    const int tid   = threadIdx.x;
    const int lane  = tid & 31;
    const int wave  = tid >> 5;
    const float* xb = xyz + (size_t)b * NPT * 3;
    for (int j = tid; j < NPT; j += 256) {
        sx[j] = xb[j * 3 + 0];
        sy[j] = xb[j * 3 + 1];
        sz[j] = xb[j * 3 + 2];
    }
    __syncthreads();

    for (int p = wave; p < 64; p += 8) {
        const int i = chunk * 64 + p;
        const float qx = sx[i], qy = sy[i], qz = sz[i];

        // per-lane sorted 8-smallest
        float d8[8];
#pragma unroll
        for (int k = 0; k < 8; ++k) d8[k] = 3.0e38f;
        for (int j = lane; j < NPT; j += 32) {
            const float dx = sx[j] - qx, dy = sy[j] - qy, dz = sz[j] - qz;
            const float d = sqrtf(fmaxf(dx * dx + dy * dy + dz * dz, 0.0f));
            if (d < d8[7]) {
                d8[7] = d;
#pragma unroll
                for (int k = 7; k > 0; --k) {
                    const float lo = fminf(d8[k - 1], d8[k]);
                    const float hi = fmaxf(d8[k - 1], d8[k]);
                    d8[k - 1] = lo;
                    d8[k]     = hi;
                }
            }
        }
        // merge 32 sorted lists: pop global min 8 times
        float kth = 0.0f;
#pragma unroll
        for (int r = 0; r < 8; ++r) {
            const float head = d8[0];
            float m = head;
#pragma unroll
            for (int s = 16; s >= 1; s >>= 1) m = fminf(m, __shfl_xor(m, s, 32));
            const unsigned long long bal = __ballot(head == m);
            const int owner = __builtin_ffsll((long long)bal) - 1;
            if (lane == owner) {
#pragma unroll
                for (int k = 0; k < 7; ++k) d8[k] = d8[k + 1];
                d8[7] = 3.0e38f;
            }
            kth = m;
        }

        // ordered selection of the first 16 zero-value indices
        int* outp = idxout + ((size_t)b * NPT + i) * NSAMP;
        int cnt = 0;
        for (int j0 = 0; j0 < NPT && cnt < NSAMP; j0 += 32) {
            const int j = j0 + lane;
            const bool inbox = (fabsf(sx[j] - qx) <= kth) &&
                               (fabsf(sy[j] - qy) <= kth) &&
                               (fabsf(sz[j] - qz) <= kth);
            const bool zv = (j == 0) || (!inbox);
            const unsigned int bm = (unsigned int)__ballot(zv);
            const int pre = __popc(bm & ((1u << lane) - 1u));
            if (zv && (cnt + pre) < NSAMP) outp[cnt + pre] = j;
            cnt += __popc(bm);
        }
        if (cnt < NSAMP) {  // fallback: in-box j>=1, ascending (values j ascending)
            int c2 = cnt;
            for (int j0 = 0; j0 < NPT && c2 < NSAMP; j0 += 32) {
                const int j = j0 + lane;
                const bool inbox = (fabsf(sx[j] - qx) <= kth) &&
                                   (fabsf(sy[j] - qy) <= kth) &&
                                   (fabsf(sz[j] - qz) <= kth);
                const bool tk = (j != 0) && inbox;
                const unsigned int bm = (unsigned int)__ballot(tk);
                const int pre = __popc(bm & ((1u << lane) - 1u));
                if (tk && (c2 + pre) < NSAMP) outp[c2 + pre] = j;
                c2 += __popc(bm);
            }
        }
    }
}

// ---------------------------------------------------------------------------
// Kernel 2: per point (one wave each): async gather, pos MLP, attention MLP,
// softmax, weighted sum. WMMA for the two big GEMMs; the scattered feature
// gather uses GLOBAL_LOAD_ASYNC_TO_LDS_B32 (ASYNCcnt) and is hidden behind
// the pos-layer1 VALU work.
// ---------------------------------------------------------------------------
struct AttnSmem {
    float gf[16][DCH];    // gathered features, f32 (A source + weighted sum)
    h16   h1[16][DCH];    // pos hidden (post-GELU), f16 A source
    h16   pos[16][DCH];   // pos output, f16 A source
    float gxyz[16][4];
    float logits[16];
    int   nidx[16];
};

__global__ __launch_bounds__(128) void attn_kernel(
    const float* __restrict__ xyz, const float* __restrict__ features,
    const float* __restrict__ pos_w1, const float* __restrict__ pos_b1,
    const h16* __restrict__ pos_w2p, const float* __restrict__ pos_b2,
    const h16* __restrict__ attn_w1p, const float* __restrict__ attn_b1,
    const float* __restrict__ attn_w2, const float* __restrict__ attn_b2,
    const int* __restrict__ idx, float* __restrict__ out_mid) {
    __shared__ AttnSmem S[4];
    const int lane = threadIdx.x & 31;
    const int wave = threadIdx.x >> 5;
    AttnSmem& s = S[wave];
    const int gpt = blockIdx.x * 4 + wave;  // global point id, exact grid
    const int b = gpt >> 12;                // / NPT
    const int i = gpt & (NPT - 1);

    if (lane < 16) s.nidx[lane] = idx[((size_t)b * NPT + i) * NSAMP + lane];
    __syncthreads();

    const float* xb = xyz + (size_t)b * NPT * 3;
    const float qx = xb[i * 3 + 0], qy = xb[i * 3 + 1], qz = xb[i * 3 + 2];
    if (lane < 16) {
        const int j = s.nidx[lane];
        s.gxyz[lane][0] = xb[j * 3 + 0] - qx;
        s.gxyz[lane][1] = xb[j * 3 + 1] - qy;
        s.gxyz[lane][2] = xb[j * 3 + 2] - qz;
    }

    // Scattered async gather (global -> LDS, no VGPR round-trip):
    //   gf[ss][c] = features[b][c][nidx[ss]]
    // LDS byte address = low 32 bits of the generic pointer (LDS aperture).
    const float* fb = features + (size_t)b * DCH * NPT;
    const unsigned gfbase = (unsigned)(uintptr_t)(&s.gf[0][0]);
    for (int t = lane; t < 16 * DCH; t += 32) {
        const int ss = t & 15, c = t >> 4;
        const float* gp = fb + (size_t)c * NPT + s.nidx[ss];
        const unsigned laddr = gfbase + (unsigned)((ss * DCH + c) << 2);
        asm volatile("global_load_async_to_lds_b32 %0, %1, off"
                     :
                     : "v"(laddr), "v"(gp)
                     : "memory");
    }

    // pos layer1 (K=3): VALU, overlapped with the async gather
    for (int t = lane; t < 16 * DCH; t += 32) {
        const int ss = t >> 8, c = t & 255;
        const float v = s.gxyz[ss][0] * pos_w1[c] +
                        s.gxyz[ss][1] * pos_w1[DCH + c] +
                        s.gxyz[ss][2] * pos_w1[2 * DCH + c] + pos_b1[c];
        s.h1[ss][c] = (h16)gelu_exact(v);
    }
    asm volatile("s_wait_asynccnt 0x0" ::: "memory");
    __syncthreads();

    const int col = lane & 15, hi = lane >> 4;

    // pos layer2: [16x256] @ [256x256] via WMMA (8 K-steps x 16 N-tiles)
    {
        v8f acc[16];
#pragma unroll
        for (int nt = 0; nt < 16; ++nt) acc[nt] = zero8();
#pragma unroll
        for (int kt = 0; kt < 8; ++kt) {
            const v16h a = a_frag_f16(&s.h1[0][0], kt * 32, lane);
#pragma unroll
            for (int nt = 0; nt < 16; ++nt) {
                const v16h bf = *(const v16h*)(pos_w2p + ((kt * 16 + nt) << 9) + (lane << 4));
                acc[nt] = wmma_f16(a, bf, acc[nt]);
            }
        }
#pragma unroll
        for (int nt = 0; nt < 16; ++nt) {
            const int c = nt * 16 + col;
            const float bias = pos_b2[c];
#pragma unroll
            for (int v = 0; v < 8; ++v) s.pos[hi * 8 + v][c] = (h16)(acc[nt][v] + bias);
        }
    }
    __syncthreads();

    // attn layer1: [16x512] @ [512x256]; fold bias+GELU+dot(attn_w2) into D frags
    {
        v8f acc[16];
#pragma unroll
        for (int nt = 0; nt < 16; ++nt) acc[nt] = zero8();
#pragma unroll
        for (int kt = 0; kt < 16; ++kt) {
            const v16h a = (kt < 8) ? a_frag_f32(&s.gf[0][0], kt * 32, lane)
                                    : a_frag_f16(&s.pos[0][0], (kt - 8) * 32, lane);
#pragma unroll
            for (int nt = 0; nt < 16; ++nt) {
                const v16h bf = *(const v16h*)(attn_w1p + ((kt * 16 + nt) << 9) + (lane << 4));
                acc[nt] = wmma_f16(a, bf, acc[nt]);
            }
        }
        float part[8];
#pragma unroll
        for (int v = 0; v < 8; ++v) part[v] = 0.0f;
#pragma unroll
        for (int nt = 0; nt < 16; ++nt) {
            const int c = nt * 16 + col;
            const float b1v = attn_b1[c];
            const float w2v = attn_w2[c];
#pragma unroll
            for (int v = 0; v < 8; ++v) part[v] += gelu_exact(acc[nt][v] + b1v) * w2v;
        }
#pragma unroll
        for (int v = 0; v < 8; ++v) {  // reduce over 16 columns (stay within half)
            float p = part[v];
            p += __shfl_xor(p, 1, 32);
            p += __shfl_xor(p, 2, 32);
            p += __shfl_xor(p, 4, 32);
            p += __shfl_xor(p, 8, 32);
            part[v] = p;
        }
        if (lane == 0 || lane == 16) {
#pragma unroll
            for (int v = 0; v < 8; ++v) s.logits[hi * 8 + v] = part[v] + attn_b2[0];
        }
    }
    __syncthreads();

    // softmax over 16 neighbors + weighted feature sum
    {
        float w[16], mx = -3.0e38f;
#pragma unroll
        for (int ss = 0; ss < 16; ++ss) mx = fmaxf(mx, s.logits[ss]);
        float sum = 0.0f;
#pragma unroll
        for (int ss = 0; ss < 16; ++ss) {
            w[ss] = expf(s.logits[ss] - mx);
            sum += w[ss];
        }
        const float inv = 1.0f / sum;
        float* ob = out_mid + (size_t)gpt * DCH;
        for (int c = lane; c < DCH; c += 32) {
            float a = 0.0f;
#pragma unroll
            for (int ss = 0; ss < 16; ++ss) a += s.gf[ss][c] * w[ss];
            ob[c] = a * inv;
        }
    }
}

// ---------------------------------------------------------------------------
// Kernel 3: FFN + residual + transpose. One wave per 16-row tile.
// ---------------------------------------------------------------------------
struct FfnSmem {
    h16   a[16][DCH];   // A fragments source (f16)
    float y[16][DCH];   // residual / output tile (f32), LDS-transposed store
};

__global__ __launch_bounds__(128) void ffn_kernel(
    const float* __restrict__ out_mid, const h16* __restrict__ w1p,
    const float* __restrict__ b1, const h16* __restrict__ w2p,
    const float* __restrict__ b2, float* __restrict__ out_final) {
    __shared__ FfnSmem S[4];
    const int lane = threadIdx.x & 31;
    const int wave = threadIdx.x >> 5;
    FfnSmem& s = S[wave];
    const int tile = blockIdx.x * 4 + wave;  // 1024 tiles
    const int g0 = tile * 16;

    for (int t = lane; t < 16 * DCH; t += 32) {
        const int r = t >> 8, c = t & 255;
        const float v = out_mid[(size_t)(g0 + r) * DCH + c];
        s.y[r][c] = v;       // residual
        s.a[r][c] = (h16)v;  // A for layer1
    }
    __syncthreads();

    const int col = lane & 15, hi = lane >> 4;

    v8f acc[16];
#pragma unroll
    for (int nt = 0; nt < 16; ++nt) acc[nt] = zero8();
#pragma unroll
    for (int kt = 0; kt < 8; ++kt) {
        const v16h a = a_frag_f16(&s.a[0][0], kt * 32, lane);
#pragma unroll
        for (int nt = 0; nt < 16; ++nt) {
            const v16h bf = *(const v16h*)(w1p + ((kt * 16 + nt) << 9) + (lane << 4));
            acc[nt] = wmma_f16(a, bf, acc[nt]);
        }
    }
    __syncthreads();
#pragma unroll
    for (int nt = 0; nt < 16; ++nt) {
        const int c = nt * 16 + col;
        const float bias = b1[c];
#pragma unroll
        for (int v = 0; v < 8; ++v)
            s.a[hi * 8 + v][c] = (h16)gelu_exact(acc[nt][v] + bias);
    }
    __syncthreads();

#pragma unroll
    for (int nt = 0; nt < 16; ++nt) acc[nt] = zero8();
#pragma unroll
    for (int kt = 0; kt < 8; ++kt) {
        const v16h a = a_frag_f16(&s.a[0][0], kt * 32, lane);
#pragma unroll
        for (int nt = 0; nt < 16; ++nt) {
            const v16h bf = *(const v16h*)(w2p + ((kt * 16 + nt) << 9) + (lane << 4));
            acc[nt] = wmma_f16(a, bf, acc[nt]);
        }
    }
#pragma unroll
    for (int nt = 0; nt < 16; ++nt) {
        const int c = nt * 16 + col;
        const float bias = b2[c];
#pragma unroll
        for (int v = 0; v < 8; ++v) s.y[hi * 8 + v][c] += acc[nt][v] + bias;
    }
    __syncthreads();

    // transposed store: out[b, c, n], rows of the tile are consecutive n
    const int bb = g0 >> 12, n0 = g0 & (NPT - 1);
    float* ob = out_final + (size_t)bb * DCH * NPT + n0;
    for (int t = lane; t < 16 * DCH; t += 32) {
        const int c = t >> 4, r = t & 15;
        ob[(size_t)c * NPT + r] = s.y[r][c];
    }
}

// ---------------------------------------------------------------------------
extern "C" void kernel_launch(void* const* d_in, const int* in_sizes, int n_in,
                              void* d_out, int out_size, void* d_ws, size_t ws_size,
                              hipStream_t stream) {
    const float* xyz     = (const float*)d_in[0];
    const float* feat    = (const float*)d_in[1];
    const float* pos_w1  = (const float*)d_in[2];
    const float* pos_b1  = (const float*)d_in[3];
    const float* pos_w2  = (const float*)d_in[4];
    const float* pos_b2  = (const float*)d_in[5];
    const float* attn_w1 = (const float*)d_in[6];
    const float* attn_b1 = (const float*)d_in[7];
    const float* attn_w2 = (const float*)d_in[8];
    const float* attn_b2 = (const float*)d_in[9];
    const float* ffn_w1  = (const float*)d_in[10];
    const float* ffn_b1  = (const float*)d_in[11];
    const float* ffn_w2  = (const float*)d_in[12];
    const float* ffn_b2  = (const float*)d_in[13];

    char* ws = (char*)d_ws;
    int*   idxbuf  = (int*)ws;                                   // 1 MB
    float* out_mid = (float*)(ws + (size_t)(1u << 20));          // 16 MB
    h16*   pw2p    = (h16*)(ws + (size_t)(17u << 20));           // packed weights
    h16*   aw1p    = pw2p + 256 * 256;
    h16*   fw1p    = aw1p + 512 * 256;
    h16*   fw2p    = fw1p + 256 * 256;

    pack_weight<<<256, 256, 0, stream>>>(pos_w2, pw2p, 256, 256);
    pack_weight<<<512, 256, 0, stream>>>(attn_w1, aw1p, 512, 256);
    pack_weight<<<256, 256, 0, stream>>>(ffn_w1, fw1p, 256, 256);
    pack_weight<<<256, 256, 0, stream>>>(ffn_w2, fw2p, 256, 256);

    knn_box_kernel<<<BQ * 64, 256, 0, stream>>>(xyz, idxbuf);

    attn_kernel<<<(BQ * NPT) / 4, 128, 0, stream>>>(
        xyz, feat, pos_w1, pos_b1, pw2p, pos_b2, aw1p, attn_b1, attn_w2, attn_b2,
        idxbuf, out_mid);

    ffn_kernel<<<(BQ * NPT) / 64, 128, 0, stream>>>(out_mid, fw1p, ffn_b1, fw2p,
                                                    ffn_b2, (float*)d_out);
}
